// MapAttention_43095701848462
// MI455X (gfx1250) — matrix operand order repaired and checked by reading, not compiled, not run on hardware
//
#include <hip/hip_runtime.h>
#include <hip/hip_bf16.h>

// ---------------------------------------------------------------------------
// MapAttention for MI455X (gfx1250): bf16 WMMA GEMMs + fp32 softmax
//   qp   = query @ W^T + b          (WMMA, reg-prefetch + LDS double buffer)
//   attn = softmax(qp @ key^T)      (WMMA, async global->LDS triple buffer)
//   out  = attn @ x                 (WMMA, reg-prefetch + LDS double buffer)
// d_out = [ out : 8*2048*768 f32 | attn : 8*2048*2048 f32 ]
// ---------------------------------------------------------------------------

typedef __attribute__((ext_vector_type(16))) __bf16 bf16x16;
typedef __attribute__((ext_vector_type(8)))  float  f32x8;
typedef __attribute__((ext_vector_type(4)))  short  s16x4;
typedef __attribute__((ext_vector_type(8)))  short  s16x8;
typedef __attribute__((ext_vector_type(16))) short  s16x16;

#define B_   8
#define NQ   2048
#define NK   2048
#define DIM  768

#define BM   128   // block tile M
#define BN   64    // block tile N
#define BK   32    // block tile K  (== WMMA K)
#define ASTR 40    // A-tile LDS row stride (elems): 32 + 8 pad  -> 80 B rows
#define BSTR 80    // B-tile LDS row stride (elems): 64 + 16 pad -> 160 B rows

// fp32 -> bf16 via the native type; backend picks hardware cvt when available
__device__ __forceinline__ short f2bf(float f) {
  return __builtin_bit_cast(short, (__bf16)f);
}

// CDNA5 async global->LDS (ASYNCcnt).  VDST holds the LDS byte address =
// low 32 bits of the flat address of the shared-memory destination.
__device__ __forceinline__ void async_ld_b128(unsigned lds_addr, const void* g) {
  asm volatile("global_load_async_to_lds_b128 %0, %1, off"
               :: "v"(lds_addr), "v"(g) : "memory");
}
__device__ __forceinline__ void wait_async_0() {
  asm volatile("s_wait_asynccnt 0x0" ::: "memory");
}
__device__ __forceinline__ void wait_async_3() {
  asm volatile("s_wait_asynccnt 0x3" ::: "memory");
}

// A operand (16x32 bf16, M x K): lane L<16 holds row M=L, K {0..7,16..23};
// lane L+16 holds row M=L, K {8..15,24..31}.  Two ds_load_b128 per operand.
__device__ __forceinline__ bf16x16 lds_load_a(const unsigned short* a,
                                              int row_base, int lane) {
  const int r = row_base + (lane & 15);
  const int h = (lane >> 4) * 8;
  const unsigned short* p = a + r * ASTR;
  s16x8 lo = *(const s16x8*)(p + h);        // K 0..7   (or 8..15)
  s16x8 hi = *(const s16x8*)(p + 16 + h);   // K 16..23 (or 24..31)
  s16x16 v;
#pragma unroll
  for (int i = 0; i < 8; ++i) { v[i] = lo[i]; v[8 + i] = hi[i]; }
  return __builtin_bit_cast(bf16x16, v);
}

// B operand (32x16 bf16, K x N): lane L<16 holds col N=L, K {0..7,16..23};
// lane L+16 holds col N=L, K {8..15,24..31}.
__device__ __forceinline__ bf16x16 lds_load_b(const unsigned short* b,
                                              int col_base, int lane) {
  const int n = col_base + (lane & 15);
  const int h = (lane >> 4) * 8;
  s16x16 v;
#pragma unroll
  for (int i = 0; i < 8; ++i) {
    v[i] = (short)b[(h + i) * BSTR + n];
    v[8 + i] = (short)b[(16 + h + i) * BSTR + n];
  }
  return __builtin_bit_cast(bf16x16, v);
}

__device__ __forceinline__ f32x8 wmma_bf16(bf16x16 a, bf16x16 b, f32x8 c) {
  return __builtin_amdgcn_wmma_f32_16x16x32_bf16(false, a, false, b,
                                                 (short)0, c, false, false);
}

// ---------------------------------------------------------------------------
// transpose + fp32->bf16 convert:  dst[c][r] = bf16(src[r][c])
// ---------------------------------------------------------------------------
__global__ __launch_bounds__(256) void transpose_f32_to_bf16(
    const float* __restrict__ src, unsigned short* __restrict__ dst,
    int R, int C) {
  size_t tid = (size_t)blockIdx.x * blockDim.x + threadIdx.x;
  if (tid >= (size_t)R * C) return;
  int r = (int)(tid % R);
  int c = (int)(tid / R);
  dst[(size_t)c * R + r] = (unsigned short)f2bf(src[(size_t)r * C + c]);
}

// ---------------------------------------------------------------------------
// Kernel 1: qp[m][n] = bf16( sum_k query[m][k] * WT[k][n] + bias[n] )
//   fp32->bf16 conversion forces reg staging; LDS double buffer, 1 barrier/step
// ---------------------------------------------------------------------------
__global__ __launch_bounds__(256) void proj_kernel(
    const float* __restrict__ query, const unsigned short* __restrict__ WT,
    const float* __restrict__ bias, unsigned short* __restrict__ qp) {
  __shared__ unsigned short Alds[2][BM * ASTR];
  __shared__ unsigned short Blds[2][BK * BSTR];
  const int t = threadIdx.x, lane = t & 31, wid = t >> 5;
  const int wm = wid >> 1, wn = wid & 1;
  const int rowBase = blockIdx.y * BM;
  const int colBase = blockIdx.x * BN;
  const int KT = DIM / BK;

  float4 aq[4];
  s16x8  bw;
  // prologue: global -> regs for stage 0
#pragma unroll
  for (int i = 0; i < 4; ++i) {
    int e = t + i * 256, r = e >> 3, c = (e & 7) * 4;
    aq[i] = *(const float4*)(query + (size_t)(rowBase + r) * DIM + c);
  }
  bw = *(const s16x8*)(WT + (size_t)(t >> 3) * DIM + colBase + (t & 7) * 8);

  f32x8 acc[2][2] = {};
  for (int kt = 0; kt < KT; ++kt) {
    const int cur = kt & 1;
    unsigned short* Ab = Alds[cur];
    unsigned short* Bb = Blds[cur];
    // regs -> LDS (convert fp32 -> bf16)
#pragma unroll
    for (int i = 0; i < 4; ++i) {
      int e = t + i * 256, r = e >> 3, c = (e & 7) * 4;
      s16x4 h;
      h[0] = f2bf(aq[i].x); h[1] = f2bf(aq[i].y);
      h[2] = f2bf(aq[i].z); h[3] = f2bf(aq[i].w);
      *(s16x4*)(Ab + r * ASTR + c) = h;
    }
    { int r = t >> 3, c = (t & 7) * 8; *(s16x8*)(Bb + r * BSTR + c) = bw; }
    // prefetch stage kt+1 into regs (in flight during WMMAs below)
    if (kt + 1 < KT) {
      const int k0 = (kt + 1) * BK;
#pragma unroll
      for (int i = 0; i < 4; ++i) {
        int e = t + i * 256, r = e >> 3, c = (e & 7) * 4;
        aq[i] = *(const float4*)(query + (size_t)(rowBase + r) * DIM + k0 + c);
      }
      bw = *(const s16x8*)(WT + (size_t)(k0 + (t >> 3)) * DIM + colBase +
                           (t & 7) * 8);
    }
    __syncthreads();
    bf16x16 a0 = lds_load_a(Ab, wm * 32, lane);
    bf16x16 a1 = lds_load_a(Ab, wm * 32 + 16, lane);
    bf16x16 b0 = lds_load_b(Bb, wn * 32, lane);
    bf16x16 b1 = lds_load_b(Bb, wn * 32 + 16, lane);
    acc[0][0] = wmma_bf16(a0, b0, acc[0][0]);
    acc[0][1] = wmma_bf16(a0, b1, acc[0][1]);
    acc[1][0] = wmma_bf16(a1, b0, acc[1][0]);
    acc[1][1] = wmma_bf16(a1, b1, acc[1][1]);
  }
  const int nloc = lane & 15, rloc = (lane >> 4) * 8;
#pragma unroll
  for (int mt = 0; mt < 2; ++mt)
#pragma unroll
    for (int nt = 0; nt < 2; ++nt) {
      int n = colBase + wn * 32 + nt * 16 + nloc;
      int m0 = rowBase + wm * 32 + mt * 16 + rloc;
      float bn = bias[n];
#pragma unroll
      for (int g = 0; g < 8; ++g)
        qp[(size_t)(m0 + g) * DIM + n] =
            (unsigned short)f2bf(acc[mt][nt][g] + bn);
    }
}

// ---------------------------------------------------------------------------
// Kernel 2: score[b][q][k] = sum_e qp[b][q][e] * keyT[e][k]   (fp32 out)
//   pure bf16 copies -> async global->LDS, triple buffer, 1 barrier/step
// ---------------------------------------------------------------------------
__global__ __launch_bounds__(256) void score_kernel(
    const unsigned short* __restrict__ qp,
    const unsigned short* __restrict__ keyT, float* __restrict__ attn) {
  __shared__ unsigned short Alds[3][BM * ASTR];
  __shared__ unsigned short Blds[3][BK * BSTR];
  const int t = threadIdx.x, lane = t & 31, wid = t >> 5;
  const int wm = wid >> 1, wn = wid & 1;
  const int rowBase = blockIdx.y * BM;   // over Nq
  const int colBase = blockIdx.x * BN;   // over Nk
  const unsigned short* A = qp + (size_t)blockIdx.z * NQ * DIM;
  float* C = attn + (size_t)blockIdx.z * NQ * NK;
  const int KT = DIM / BK;

  // per-thread addresses for one stage (3 async b128 ops: 2 for A, 1 for B)
  const int ra0 = (t + 0) >> 2,   ca0 = ((t + 0) & 3) * 8;
  const int ra1 = (t + 256) >> 2, ca1 = ((t + 256) & 3) * 8;
  const int rb  = t >> 3,         cb  = (t & 7) * 8;

  // prologue: async stage 0 -> buffer 0
  {
    async_ld_b128((unsigned)(uintptr_t)(Alds[0] + ra0 * ASTR + ca0),
                  A + (size_t)(rowBase + ra0) * DIM + ca0);
    async_ld_b128((unsigned)(uintptr_t)(Alds[0] + ra1 * ASTR + ca1),
                  A + (size_t)(rowBase + ra1) * DIM + ca1);
    async_ld_b128((unsigned)(uintptr_t)(Blds[0] + rb * BSTR + cb),
                  keyT + (size_t)rb * NK + colBase + cb);
  }

  f32x8 acc[2][2] = {};
  int cur = 0;
  for (int kt = 0; kt < KT; ++kt) {
    const int nxt = (cur == 2) ? 0 : cur + 1;
    if (kt + 1 < KT) {
      const int k0 = (kt + 1) * BK;
      async_ld_b128((unsigned)(uintptr_t)(Alds[nxt] + ra0 * ASTR + ca0),
                    A + (size_t)(rowBase + ra0) * DIM + k0 + ca0);
      async_ld_b128((unsigned)(uintptr_t)(Alds[nxt] + ra1 * ASTR + ca1),
                    A + (size_t)(rowBase + ra1) * DIM + k0 + ca1);
      async_ld_b128((unsigned)(uintptr_t)(Blds[nxt] + rb * BSTR + cb),
                    keyT + (size_t)(k0 + rb) * NK + colBase + cb);
      wait_async_3();   // stage kt landed; stage kt+1 still in flight
    } else {
      wait_async_0();
    }
    __syncthreads();
    bf16x16 a0 = lds_load_a(Alds[cur], wm * 32, lane);
    bf16x16 a1 = lds_load_a(Alds[cur], wm * 32 + 16, lane);
    bf16x16 b0 = lds_load_b(Blds[cur], wn * 32, lane);
    bf16x16 b1 = lds_load_b(Blds[cur], wn * 32 + 16, lane);
    acc[0][0] = wmma_bf16(a0, b0, acc[0][0]);
    acc[0][1] = wmma_bf16(a0, b1, acc[0][1]);
    acc[1][0] = wmma_bf16(a1, b0, acc[1][0]);
    acc[1][1] = wmma_bf16(a1, b1, acc[1][1]);
    cur = nxt;
  }
  const int nloc = lane & 15, rloc = (lane >> 4) * 8;
#pragma unroll
  for (int mt = 0; mt < 2; ++mt)
#pragma unroll
    for (int nt = 0; nt < 2; ++nt) {
      int n = colBase + wn * 32 + nt * 16 + nloc;
      int m0 = rowBase + wm * 32 + mt * 16 + rloc;
#pragma unroll
      for (int g = 0; g < 8; ++g)
        C[(size_t)(m0 + g) * NK + n] = acc[mt][nt][g];
    }
}

// ---------------------------------------------------------------------------
// Kernel 3: in-place row softmax over NK=2048 (one block per row)
// ---------------------------------------------------------------------------
__global__ __launch_bounds__(256) void softmax_kernel(float* __restrict__ attn) {
  const int t = threadIdx.x, lane = t & 31, wid = t >> 5;
  float* p = attn + (size_t)blockIdx.x * NK;
  float v[8];
#pragma unroll
  for (int i = 0; i < 8; ++i) v[i] = p[t + i * 256];

  float m = v[0];
#pragma unroll
  for (int i = 1; i < 8; ++i) m = fmaxf(m, v[i]);
#pragma unroll
  for (int off = 16; off > 0; off >>= 1) m = fmaxf(m, __shfl_xor(m, off, 32));

  __shared__ float red[8];
  __shared__ float bc;
  if (lane == 0) red[wid] = m;
  __syncthreads();
  if (t == 0) {
    float mm = red[0];
#pragma unroll
    for (int i = 1; i < 8; ++i) mm = fmaxf(mm, red[i]);
    bc = mm;
  }
  __syncthreads();
  m = bc;

  float s = 0.f;
#pragma unroll
  for (int i = 0; i < 8; ++i) { v[i] = __expf(v[i] - m); s += v[i]; }
#pragma unroll
  for (int off = 16; off > 0; off >>= 1) s += __shfl_xor(s, off, 32);
  if (lane == 0) red[wid] = s;
  __syncthreads();
  if (t == 0) {
    float ss = red[0];
#pragma unroll
    for (int i = 1; i < 8; ++i) ss += red[i];
    bc = 1.0f / ss;
  }
  __syncthreads();
  const float inv = bc;
#pragma unroll
  for (int i = 0; i < 8; ++i) p[t + i * 256] = v[i] * inv;
}

// ---------------------------------------------------------------------------
// Kernel 4: out[b][q][d] = sum_k attn[b][q][k] * x[b][k][d]
//   fp32->bf16 conversion -> reg staging; LDS double buffer, 1 barrier/step
// ---------------------------------------------------------------------------
__global__ __launch_bounds__(256) void out_kernel(
    const float* __restrict__ attn, const float* __restrict__ x,
    float* __restrict__ out) {
  __shared__ unsigned short Alds[2][BM * ASTR];
  __shared__ unsigned short Blds[2][BK * BSTR];
  const int t = threadIdx.x, lane = t & 31, wid = t >> 5;
  const int wm = wid >> 1, wn = wid & 1;
  const int rowBase = blockIdx.y * BM;   // over Nq
  const int colBase = blockIdx.x * BN;   // over D
  const float* A = attn + (size_t)blockIdx.z * NQ * NK;
  const float* Bsrc = x + (size_t)blockIdx.z * NK * DIM;
  float* C = out + (size_t)blockIdx.z * NQ * DIM;
  const int KT = NK / BK;

  float4 aA[4], bX[2];
#pragma unroll
  for (int i = 0; i < 4; ++i) {
    int e = t + i * 256, r = e >> 3, c = (e & 7) * 4;
    aA[i] = *(const float4*)(A + (size_t)(rowBase + r) * NK + c);
  }
#pragma unroll
  for (int i = 0; i < 2; ++i) {
    int e = t + i * 256, r = e >> 4, c = (e & 15) * 4;
    bX[i] = *(const float4*)(Bsrc + (size_t)r * DIM + colBase + c);
  }

  f32x8 acc[2][2] = {};
  for (int kt = 0; kt < KT; ++kt) {
    const int cur = kt & 1;
    unsigned short* Ab = Alds[cur];
    unsigned short* Bb = Blds[cur];
#pragma unroll
    for (int i = 0; i < 4; ++i) {
      int e = t + i * 256, r = e >> 3, c = (e & 7) * 4;
      s16x4 h;
      h[0] = f2bf(aA[i].x); h[1] = f2bf(aA[i].y);
      h[2] = f2bf(aA[i].z); h[3] = f2bf(aA[i].w);
      *(s16x4*)(Ab + r * ASTR + c) = h;
    }
#pragma unroll
    for (int i = 0; i < 2; ++i) {
      int e = t + i * 256, r = e >> 4, c = (e & 15) * 4;
      s16x4 h;
      h[0] = f2bf(bX[i].x); h[1] = f2bf(bX[i].y);
      h[2] = f2bf(bX[i].z); h[3] = f2bf(bX[i].w);
      *(s16x4*)(Bb + r * BSTR + c) = h;
    }
    if (kt + 1 < KT) {
      const int k0 = (kt + 1) * BK;
#pragma unroll
      for (int i = 0; i < 4; ++i) {
        int e = t + i * 256, r = e >> 3, c = (e & 7) * 4;
        aA[i] = *(const float4*)(A + (size_t)(rowBase + r) * NK + k0 + c);
      }
#pragma unroll
      for (int i = 0; i < 2; ++i) {
        int e = t + i * 256, r = e >> 4, c = (e & 15) * 4;
        bX[i] = *(const float4*)(Bsrc + (size_t)(k0 + r) * DIM + colBase + c);
      }
    }
    __syncthreads();
    bf16x16 a0 = lds_load_a(Ab, wm * 32, lane);
    bf16x16 a1 = lds_load_a(Ab, wm * 32 + 16, lane);
    bf16x16 b0 = lds_load_b(Bb, wn * 32, lane);
    bf16x16 b1 = lds_load_b(Bb, wn * 32 + 16, lane);
    acc[0][0] = wmma_bf16(a0, b0, acc[0][0]);
    acc[0][1] = wmma_bf16(a0, b1, acc[0][1]);
    acc[1][0] = wmma_bf16(a1, b0, acc[1][0]);
    acc[1][1] = wmma_bf16(a1, b1, acc[1][1]);
  }
  const int nloc = lane & 15, rloc = (lane >> 4) * 8;
#pragma unroll
  for (int mt = 0; mt < 2; ++mt)
#pragma unroll
    for (int nt = 0; nt < 2; ++nt) {
      int n = colBase + wn * 32 + nt * 16 + nloc;
      int m0 = rowBase + wm * 32 + mt * 16 + rloc;
#pragma unroll
      for (int g = 0; g < 8; ++g)
        C[(size_t)(m0 + g) * DIM + n] = acc[mt][nt][g];
    }
}

// ---------------------------------------------------------------------------
extern "C" void kernel_launch(void* const* d_in, const int* in_sizes, int n_in,
                              void* d_out, int out_size, void* d_ws,
                              size_t ws_size, hipStream_t stream) {
  (void)in_sizes; (void)n_in; (void)out_size; (void)ws_size;
  const float* query = (const float*)d_in[0];   // [8,2048,768]
  const float* key   = (const float*)d_in[1];   // [2048,768]
  const float* x     = (const float*)d_in[2];   // [8,2048,768]
  const float* W     = (const float*)d_in[3];   // [768,768]
  const float* bias  = (const float*)d_in[4];   // [768]

  float* out  = (float*)d_out;                          // 8*2048*768
  float* attn = out + (size_t)B_ * NQ * DIM;            // 8*2048*2048

  // workspace: qp bf16 [16384][768] | keyT bf16 [768][2048] | WT bf16 [768][768]
  unsigned short* qp   = (unsigned short*)d_ws;
  unsigned short* keyT = qp + (size_t)B_ * NQ * DIM;
  unsigned short* WT   = keyT + (size_t)DIM * NK;

  {
    int total = DIM * DIM;
    transpose_f32_to_bf16<<<(total + 255) / 256, 256, 0, stream>>>(W, WT, DIM, DIM);
  }
  {
    int total = NK * DIM;
    transpose_f32_to_bf16<<<(total + 255) / 256, 256, 0, stream>>>(key, keyT, NK, DIM);
  }
  proj_kernel<<<dim3(DIM / BN, (B_ * NQ) / BM), 256, 0, stream>>>(query, WT, bias, qp);
  score_kernel<<<dim3(NK / BN, NQ / BM, B_), 256, 0, stream>>>(qp, keyT, attn);
  softmax_kernel<<<dim3(B_ * NQ), 256, 0, stream>>>(attn);
  out_kernel<<<dim3(DIM / BN, NQ / BM, B_), 256, 0, stream>>>(attn, x, out);
}
